// LinearDecoderV2_55929064128633
// MI455X (gfx1250) — compile-verified, matrix operand order checked by reading, not run on hardware
//
#include <hip/hip_runtime.h>
#include <hip/hip_bf16.h>

#define USE_ASYNC_LDS 1

// ---------------------------------------------------------------------------
// Types for CDNA5 WMMA bf16 (wave32): A/B = v16bf (8 VGPRs), C/D = v8f.
// ---------------------------------------------------------------------------
typedef __attribute__((ext_vector_type(16))) __bf16 v16bf;
typedef __attribute__((ext_vector_type(8)))  float  v8f;

union FragU {
    v16bf v;
    uint4 u[2];
};

__device__ __forceinline__ unsigned short f2bf(float f) {
    unsigned int u = __float_as_uint(f);
    unsigned int r = (u + 0x7FFFu + ((u >> 16) & 1u)) >> 16;   // RNE
    return (unsigned short)r;
}

__device__ __forceinline__ float sigmoidf(float x) {
    return 1.0f / (1.0f + expf(-x));
}

#if __HIP_DEVICE_COMPILE__
#define WMMA_BF16(a, b, c) __builtin_amdgcn_wmma_f32_16x16x32_bf16(false, (a), false, (b), (short)0, (c), false, false)
#else
#define WMMA_BF16(a, b, c) (c)
#endif

// Async 16-byte global -> LDS copy (ASYNCcnt-tracked Tensor/async path).
// VDST VGPR holds the LDS byte address (low 32 bits of the generic pointer),
// VADDR the 64-bit global address. EXEC is all-ones at every call site.
__device__ __forceinline__ void async_copy16(const void* gsrc, unsigned short* ldst) {
#if USE_ASYNC_LDS && __HIP_DEVICE_COMPILE__
    unsigned lofs = (unsigned)(size_t)ldst;   // LDS aperture: low 32 bits = LDS offset
    asm volatile("global_load_async_to_lds_b128 %0, %1, off"
                 :: "v"(lofs), "v"(gsrc) : "memory");
#else
    *(uint4*)ldst = *(const uint4*)gsrc;      // fallback: vmem load + ds_store_b128
#endif
}

__device__ __forceinline__ void wait_async() {
#if USE_ASYNC_LDS && __HIP_DEVICE_COMPILE__
    asm volatile("s_wait_asynccnt 0" ::: "memory");
#endif
}

// ---------------------------------------------------------------------------
// bf16 GEMM: out[M,N] = act(A[M,K] * W[N,K]^T + bias), v_wmma_f32_16x16x32_bf16.
// LDS-staged, double-buffered pipeline:
//   K is processed in 32-wide stages. Per stage the block async-copies the
//   128x32 A-slab and 128x32 B-slab into LDS (256 threads x 2 chunks x 16B
//   per matrix), ping-pong buffered; s_wait_asynccnt + barrier publish a
//   buffer while WMMAs consume the other from ds_load_b128.
// Per wave: 32(M) x 64(N) tile = 8 WMMAs/stage from 2 A-frags x 4 B-frags.
// Block = 256 threads = 8 waves arranged 4(M) x 2(N) -> 128x128 per block.
// Grid = (N/128, M/128). Exact tiling; EXEC all-ones (WMMA requirement).
// LDS rows padded to 40 elems (80B) to spread banks; 2x2 buffers = 40KB.
// ---------------------------------------------------------------------------
#define LDS_LD 40   // 32 K-elements + 8 pad (80 bytes, 16B aligned)

__global__ __launch_bounds__(256) void gemm_bf16_wmma(
    const unsigned short* __restrict__ A, int lda,
    const unsigned short* __restrict__ W, int ldb,
    const float* __restrict__ bias,
    void* __restrict__ out, int ldo,
    int K, int act, int outBf16)
{
    __shared__ unsigned short lsA[2][128 * LDS_LD];
    __shared__ unsigned short lsB[2][128 * LDS_LD];

    const int tid  = threadIdx.x;
    const int lane = tid & 31;
    const int wave = tid >> 5;
    const int r    = lane & 15;          // row within 16
    const int half = lane >> 4;          // K-half select (ISA 16-bit A/B layout)
    const int bm0  = blockIdx.y * 128;   // block row base (A / out rows)
    const int bn0  = blockIdx.x * 128;   // block col base (W rows / out cols)
    const int wm   = (wave & 3) * 32;    // wave tile within block (M)
    const int wn   = (wave >> 2) * 64;   // wave tile within block (N)

    // Per-thread copy assignment: 512 16B-chunks per matrix per stage
    // (128 rows x 4 chunks); thread covers chunks {tid, tid+256}.
    const int c0row = tid >> 2, c0kc = (tid & 3) * 8;
    const int c1row = (tid + 256) >> 2, c1kc = ((tid + 256) & 3) * 8;

    const unsigned short* Ag = A + (size_t)bm0 * lda;
    const unsigned short* Wg = W + (size_t)bn0 * ldb;

    auto stage = [&](int buf, int k0) {
        async_copy16(Ag + (size_t)c0row * lda + k0 + c0kc, &lsA[buf][c0row * LDS_LD + c0kc]);
        async_copy16(Ag + (size_t)c1row * lda + k0 + c1kc, &lsA[buf][c1row * LDS_LD + c1kc]);
        async_copy16(Wg + (size_t)c0row * ldb + k0 + c0kc, &lsB[buf][c0row * LDS_LD + c0kc]);
        async_copy16(Wg + (size_t)c1row * ldb + k0 + c1kc, &lsB[buf][c1row * LDS_LD + c1kc]);
    };

    v8f acc[2][4] = {};
    const int S = K >> 5;                // stages of 32

    stage(0, 0);
    wait_async();
    __syncthreads();

    for (int s = 0; s < S; ++s) {
        const int cur = s & 1;
        if (s + 1 < S) stage(cur ^ 1, (s + 1) << 5);

        const unsigned short* As = &lsA[cur][0];
        const unsigned short* Bs = &lsB[cur][0];
        FragU a0, a1, b0, b1, b2, b3;
        const unsigned short* ar0 = As + (wm + r) * LDS_LD + half * 8;
        const unsigned short* ar1 = ar0 + 16 * LDS_LD;
        const unsigned short* br0 = Bs + (wn + r) * LDS_LD + half * 8;
        a0.u[0] = *(const uint4*)(ar0);                 a0.u[1] = *(const uint4*)(ar0 + 16);
        a1.u[0] = *(const uint4*)(ar1);                 a1.u[1] = *(const uint4*)(ar1 + 16);
        b0.u[0] = *(const uint4*)(br0);                 b0.u[1] = *(const uint4*)(br0 + 16);
        b1.u[0] = *(const uint4*)(br0 + 16 * LDS_LD);   b1.u[1] = *(const uint4*)(br0 + 16 * LDS_LD + 16);
        b2.u[0] = *(const uint4*)(br0 + 32 * LDS_LD);   b2.u[1] = *(const uint4*)(br0 + 32 * LDS_LD + 16);
        b3.u[0] = *(const uint4*)(br0 + 48 * LDS_LD);   b3.u[1] = *(const uint4*)(br0 + 48 * LDS_LD + 16);

        acc[0][0] = WMMA_BF16(a0.v, b0.v, acc[0][0]);
        acc[0][1] = WMMA_BF16(a0.v, b1.v, acc[0][1]);
        acc[0][2] = WMMA_BF16(a0.v, b2.v, acc[0][2]);
        acc[0][3] = WMMA_BF16(a0.v, b3.v, acc[0][3]);
        acc[1][0] = WMMA_BF16(a1.v, b0.v, acc[1][0]);
        acc[1][1] = WMMA_BF16(a1.v, b1.v, acc[1][1]);
        acc[1][2] = WMMA_BF16(a1.v, b2.v, acc[1][2]);
        acc[1][3] = WMMA_BF16(a1.v, b3.v, acc[1][3]);

        if (s + 1 < S) {
            wait_async();        // own async copies for next stage complete
            __syncthreads();     // publish next buffer / release current one
        }
    }

    // C/D layout: VGPR rr -> M = base + rr + 8*half ; N = base + 16*t + (lane&15)
#pragma unroll
    for (int i = 0; i < 2; ++i) {
#pragma unroll
        for (int t = 0; t < 4; ++t) {
#pragma unroll
            for (int rr = 0; rr < 8; ++rr) {
                int row = bm0 + wm + i * 16 + rr + half * 8;
                int col = bn0 + wn + t * 16 + r;
                float v = acc[i][t][rr];
                if (bias) v += bias[col];
                if (act == 1) v = (v >= 0.f) ? v : 0.2f * v;   // LeakyReLU(0.2)
                if (outBf16) ((unsigned short*)out)[(size_t)row * ldo + col] = f2bf(v);
                else         ((float*)out)[(size_t)row * ldo + col] = v;
            }
        }
    }
}

// ---------------------------------------------------------------------------
// Elementwise / conversion helpers
// ---------------------------------------------------------------------------
__global__ void conv_bf16_strided(const float* __restrict__ src, unsigned short* __restrict__ dst,
                                  int rows, int cols, int dst_ld) {
    int idx = blockIdx.x * blockDim.x + threadIdx.x;
    if (idx >= rows * cols) return;
    int rr = idx / cols, cc = idx - rr * cols;
    dst[(size_t)rr * dst_ld + cc] = f2bf(src[idx]);
}

__global__ void add2_bf16(const float* __restrict__ a, const float* __restrict__ b,
                          unsigned short* __restrict__ dst, int n) {
    int idx = blockIdx.x * blockDim.x + threadIdx.x;
    if (idx < n) dst[idx] = f2bf(a[idx] + b[idx]);
}

__global__ void zero_f32(float* p, int n) {
    int idx = blockIdx.x * blockDim.x + threadIdx.x;
    if (idx < n) p[idx] = 0.0f;
}

__global__ void zero_bf16_strided(unsigned short* dst, int rows, int cols, int ld) {
    int idx = blockIdx.x * blockDim.x + threadIdx.x;
    if (idx >= rows * cols) return;
    int rr = idx / cols, cc = idx - rr * cols;
    dst[(size_t)rr * ld + cc] = 0;
}

// Decode LSTM pointwise: gates[1024,1024] (i|f|g|o, 256 each), c in/out f32,
// h -> bf16 into Ap[:,0:256] (ld 288).
__global__ void dec_pointwise(const float* __restrict__ gates,
                              const float* __restrict__ bih, const float* __restrict__ bhh,
                              float* __restrict__ c, unsigned short* __restrict__ Ap) {
    int idx = blockIdx.x * blockDim.x + threadIdx.x;   // < 1024*256
    int b = idx >> 8;
    int j = idx & 255;
    const float* g = gates + (size_t)b * 1024;
    float ig = sigmoidf(g[j]       + bih[j]       + bhh[j]);
    float fg = sigmoidf(g[256 + j] + bih[256 + j] + bhh[256 + j]);
    float gg = tanhf   (g[512 + j] + bih[512 + j] + bhh[512 + j]);
    float og = sigmoidf(g[768 + j] + bih[768 + j] + bhh[768 + j]);
    float cn = fg * c[idx] + ig * gg;
    c[idx] = cn;
    float hn = og * tanhf(cn);
    Ap[(size_t)b * 288 + j] = f2bf(hn);
}

// Predictor LSTM pointwise + fused softmax/sigmoid output.
// Block = 256 = 8 batch rows x 32 lanes (I=32 = one wave row on wave32).
__global__ void pred_pointwise(const float* __restrict__ pg,
                               const float* __restrict__ bih, const float* __restrict__ bhh,
                               float* __restrict__ pc, unsigned short* __restrict__ Ap,
                               float* __restrict__ out, int l) {
    int lane = threadIdx.x & 31;
    int b = blockIdx.x * 8 + (threadIdx.x >> 5);
    const float* g = pg + (size_t)b * 128;
    int j = lane;
    float ig = sigmoidf(g[j]      + bih[j]      + bhh[j]);
    float fg = sigmoidf(g[32 + j] + bih[32 + j] + bhh[32 + j]);
    float gg = tanhf   (g[64 + j] + bih[64 + j] + bhh[64 + j]);
    float og = sigmoidf(g[96 + j] + bih[96 + j] + bhh[96 + j]);
    int ci = b * 32 + j;
    float cn = fg * pc[ci] + ig * gg;
    pc[ci] = cn;
    float hn = og * tanhf(cn);
    Ap[(size_t)b * 288 + 256 + j] = f2bf(hn);

    // softmax over lanes 0..30, sigmoid on lane 31 (wave32 butterfly)
    float vmax = (lane < 31) ? hn : -3.0e38f;
    for (int m = 16; m >= 1; m >>= 1) vmax = fmaxf(vmax, __shfl_xor(vmax, m, 32));
    float e = (lane < 31) ? expf(hn - vmax) : 0.0f;
    float s = e;
    for (int m = 16; m >= 1; m >>= 1) s += __shfl_xor(s, m, 32);
    float res = (lane < 31) ? (e / s) : sigmoidf(hn);
    out[((size_t)b * 128 + l) * 32 + j] = res;   // [B, L, I]
}

// ---------------------------------------------------------------------------
// Host orchestration
// ---------------------------------------------------------------------------
extern "C" void kernel_launch(void* const* d_in, const int* in_sizes, int n_in,
                              void* d_out, int out_size, void* d_ws, size_t ws_size,
                              hipStream_t stream) {
    (void)in_sizes; (void)n_in; (void)out_size; (void)ws_size;

    const float* x    = (const float*)d_in[0];
    const float* W1   = (const float*)d_in[1];  const float* b1  = (const float*)d_in[2];
    const float* W2   = (const float*)d_in[3];  const float* b2  = (const float*)d_in[4];
    const float* W3   = (const float*)d_in[5];  const float* b3  = (const float*)d_in[6];
    const float* Wh1  = (const float*)d_in[7];  const float* bh1 = (const float*)d_in[8];
    const float* Wh2  = (const float*)d_in[9];  const float* bh2 = (const float*)d_in[10];
    const float* Wc1  = (const float*)d_in[11]; const float* bc1 = (const float*)d_in[12];
    const float* Wc2  = (const float*)d_in[13]; const float* bc2 = (const float*)d_in[14];
    const float* Wx1  = (const float*)d_in[15]; const float* bx1 = (const float*)d_in[16];
    const float* Wx2  = (const float*)d_in[17]; const float* bx2 = (const float*)d_in[18];
    const float* rWih = (const float*)d_in[19]; const float* rWhh = (const float*)d_in[20];
    const float* rbih = (const float*)d_in[21]; const float* rbhh = (const float*)d_in[22];
    const float* pWih = (const float*)d_in[23]; const float* pWhh = (const float*)d_in[24];
    const float* pbih = (const float*)d_in[25]; const float* pbhh = (const float*)d_in[26];
    float* out = (float*)d_out;

    char* ws = (char*)d_ws;
    size_t off = 0;
    auto alloc = [&](size_t bytes) -> void* {
        size_t cur = off;
        off = (off + bytes + 255) & ~(size_t)255;
        return (void*)(ws + cur);
    };

    // bf16 buffers
    unsigned short* W1b   = (unsigned short*)alloc((size_t)512 * 128 * 2);
    unsigned short* W2b   = (unsigned short*)alloc((size_t)512 * 512 * 2);
    unsigned short* W3b   = (unsigned short*)alloc((size_t)512 * 512 * 2);
    unsigned short* Wh1b  = (unsigned short*)alloc((size_t)512 * 512 * 2);
    unsigned short* Wh2b  = (unsigned short*)alloc((size_t)256 * 512 * 2);
    unsigned short* Wc1b  = (unsigned short*)alloc((size_t)512 * 512 * 2);
    unsigned short* Wc2b  = (unsigned short*)alloc((size_t)256 * 512 * 2);
    unsigned short* Wx1b  = (unsigned short*)alloc((size_t)512 * 512 * 2);
    unsigned short* Wx2b  = (unsigned short*)alloc((size_t)256 * 512 * 2);
    unsigned short* rWcat = (unsigned short*)alloc((size_t)1024 * 512 * 2);  // [rWih|rWhh] along K
    unsigned short* rWsum = (unsigned short*)alloc((size_t)1024 * 256 * 2);  // rWih + rWhh
    unsigned short* pWcat = (unsigned short*)alloc((size_t)128 * 288 * 2);   // [pWih|pWhh] along K
    unsigned short* xb    = (unsigned short*)alloc((size_t)1024 * 128 * 2);
    unsigned short* t1    = (unsigned short*)alloc((size_t)1024 * 512 * 2);
    unsigned short* t2    = (unsigned short*)alloc((size_t)1024 * 512 * 2);
    unsigned short* t3    = (unsigned short*)alloc((size_t)1024 * 512 * 2);
    unsigned short* ubuf  = (unsigned short*)alloc((size_t)1024 * 512 * 2);
    unsigned short* A1    = (unsigned short*)alloc((size_t)1024 * 512 * 2);  // [x0 | h0]
    unsigned short* Ap    = (unsigned short*)alloc((size_t)1024 * 288 * 2);  // [dec_h | pred_h]
    // f32 buffers
    float* gates  = (float*)alloc((size_t)1024 * 1024 * 4);
    float* pgates = (float*)alloc((size_t)1024 * 128 * 4);
    float* cbuf   = (float*)alloc((size_t)1024 * 256 * 4);
    float* pcbuf  = (float*)alloc((size_t)1024 * 32 * 4);

    auto conv = [&](const float* src, unsigned short* dst, int rows, int cols, int dld) {
        int n = rows * cols;
        conv_bf16_strided<<<(n + 255) / 256, 256, 0, stream>>>(src, dst, rows, cols, dld);
    };
    auto gemm = [&](const unsigned short* A, int lda, const unsigned short* Wm, int ldb,
                    const float* bias, void* o, int ldo, int M, int N, int K,
                    int act, int obf) {
        dim3 grid(N / 128, M / 128);
        gemm_bf16_wmma<<<grid, 256, 0, stream>>>(A, lda, Wm, ldb, bias, o, ldo, K, act, obf);
    };

    // ---- Weight conversion / preprocessing (deterministic every call) ----
    conv(W1, W1b, 512, 128, 128);
    conv(W2, W2b, 512, 512, 512);
    conv(W3, W3b, 512, 512, 512);
    conv(Wh1, Wh1b, 512, 512, 512);
    conv(Wh2, Wh2b, 256, 512, 512);
    conv(Wc1, Wc1b, 512, 512, 512);
    conv(Wc2, Wc2b, 256, 512, 512);
    conv(Wx1, Wx1b, 512, 512, 512);
    conv(Wx2, Wx2b, 256, 512, 512);
    conv(rWih, rWcat, 1024, 256, 512);
    conv(rWhh, rWcat + 256, 1024, 256, 512);
    add2_bf16<<<(1024 * 256 + 255) / 256, 256, 0, stream>>>(rWih, rWhh, rWsum, 1024 * 256);
    conv(pWih, pWcat, 128, 256, 288);
    conv(pWhh, pWcat + 256, 128, 32, 288);
    conv(x, xb, 1024, 128, 128);
    zero_f32<<<(1024 * 32 + 255) / 256, 256, 0, stream>>>(pcbuf, 1024 * 32);
    zero_bf16_strided<<<(1024 * 32 + 255) / 256, 256, 0, stream>>>(Ap + 256, 1024, 32, 288);

    // ---- MLP front-end (bf16 WMMA, LeakyReLU fused) ----
    gemm(xb, 128, W1b, 128, b1, t1, 512, 1024, 512, 128, 1, 1);
    gemm(t1, 512, W2b, 512, b2, t2, 512, 1024, 512, 512, 1, 1);
    gemm(t2, 512, W3b, 512, b3, t3, 512, 1024, 512, 512, 1, 1);
    // h0 head -> A1[:,256:512] (bf16)
    gemm(t3, 512, Wh1b, 512, bh1, ubuf, 512, 1024, 512, 512, 1, 1);
    gemm(ubuf, 512, Wh2b, 512, bh2, A1 + 256, 512, 1024, 256, 512, 0, 1);
    // c0 head -> cbuf (f32)
    gemm(t3, 512, Wc1b, 512, bc1, ubuf, 512, 1024, 512, 512, 1, 1);
    gemm(ubuf, 512, Wc2b, 512, bc2, cbuf, 256, 1024, 256, 512, 0, 0);
    // x0 head -> A1[:,0:256] (bf16)
    gemm(t3, 512, Wx1b, 512, bx1, ubuf, 512, 1024, 512, 512, 1, 1);
    gemm(ubuf, 512, Wx2b, 512, bx2, A1, 512, 1024, 256, 512, 0, 1);

    // ---- Interleaved decode + predictor scans (128 sequential steps) ----
    for (int l = 0; l < 128; ++l) {
        if (l == 0) {
            // gates = [x0|h0] @ [rWih|rWhh]^T
            gemm(A1, 512, rWcat, 512, nullptr, gates, 1024, 1024, 1024, 512, 0, 0);
        } else {
            // tok == h after step 1: gates = h @ (rWih + rWhh)^T
            gemm(Ap, 288, rWsum, 256, nullptr, gates, 1024, 1024, 1024, 256, 0, 0);
        }
        dec_pointwise<<<(1024 * 256) / 256, 256, 0, stream>>>(gates, rbih, rbhh, cbuf, Ap);
        // predictor consumes this step's decode output: pgates = [h_dec|h_pred] @ [pWih|pWhh]^T
        gemm(Ap, 288, pWcat, 288, nullptr, pgates, 128, 1024, 128, 288, 0, 0);
        pred_pointwise<<<1024 / 8, 256, 0, stream>>>(pgates, pbih, pbhh, pcbuf, Ap, out, l);
    }
}
